// CrossGraphModel_25872882991238
// MI455X (gfx1250) — compile-verified, matrix operand order checked by reading, not run on hardware
//
#include <hip/hip_runtime.h>
#include <hip/hip_bf16.h>
#include <math.h>

#define NN   32768
#define EE   262144
#define BB   64
#define DIN  20
#define HH   512
#define BN_EPS 1e-5f

typedef __attribute__((ext_vector_type(16))) __bf16 v16bf;
typedef __attribute__((ext_vector_type(8)))  __bf16 v8bf;
typedef __attribute__((ext_vector_type(8)))  float  v8f;

// ---------------------------------------------------------------- utilities
__global__ void fill_zero_kernel(float* p, long n) {
    long i = (long)blockIdx.x * blockDim.x + threadIdx.x;
    if (i < n) p[i] = 0.0f;
}

// ------------------------------------------------- h0 = x @ W_init + b_init
__global__ void init_linear_kernel(const float* __restrict__ x,
                                   const float* __restrict__ W,   // [DIN,HH]
                                   const float* __restrict__ b,
                                   float* __restrict__ out) {
    long i = (long)blockIdx.x * blockDim.x + threadIdx.x;
    if (i >= (long)NN * HH) return;
    int n = (int)(i / HH), j = (int)(i % HH);
    const float* xr = x + (long)n * DIN;
    float s = b[j];
    #pragma unroll
    for (int d = 0; d < DIN; ++d) s += xr[d] * W[d * HH + j];
    out[i] = s;
}

// ----------------------------------------- weight fp32 [K,N] -> bf16 [N,K]
__global__ void transpose_w_bf16_kernel(const float* __restrict__ W,
                                        __bf16* __restrict__ Wt) {
    long i = (long)blockIdx.x * blockDim.x + threadIdx.x;
    if (i >= (long)HH * HH) return;
    int k = (int)(i / HH), n = (int)(i % HH);
    Wt[(long)n * HH + k] = (__bf16)W[(long)k * HH + n];
}

// ------------------------------------------------------------- degree count
__global__ void deg_kernel(const int* __restrict__ dst, float* __restrict__ deg) {
    int e = blockIdx.x * blockDim.x + threadIdx.x;
    if (e < EE) atomicAdd(&deg[dst[e]], 1.0f);
}

// ----------------------------------------- agg[dst] += h[src]  (per feature)
__global__ void scatter_kernel(const float* __restrict__ h,
                               const int* __restrict__ src,
                               const int* __restrict__ dst,
                               float* __restrict__ agg) {
    long t = (long)blockIdx.x * blockDim.x + threadIdx.x;
    const int chunks = HH / 4;                       // 128 float4 chunks
    if (t >= (long)EE * chunks) return;
    int e = (int)(t / chunks);
    int c = (int)(t % chunks);
    const float4 v = *(const float4*)(h + (long)src[e] * HH + c * 4);
    float* a = agg + (long)dst[e] * HH + c * 4;
    atomicAdd(a + 0, v.x);
    atomicAdd(a + 1, v.y);
    atomicAdd(a + 2, v.z);
    atomicAdd(a + 3, v.w);
}

// --------------------------------- u_bf16 = bf16( h + agg / max(deg,1) )
__global__ void combine_bf16_kernel(const float* __restrict__ h,
                                    const float* __restrict__ agg,
                                    const float* __restrict__ deg,
                                    __bf16* __restrict__ ubf) {
    long i = (long)blockIdx.x * blockDim.x + threadIdx.x;
    if (i >= (long)NN * HH) return;
    int n = (int)(i / HH);
    float u = h[i] + agg[i] / fmaxf(deg[n], 1.0f);
    ubf[i] = (__bf16)u;
}

// ------------------------------------------------ WMMA GEMM  C = A*Bt + b
// A:  [M x 512] bf16 row-major      Bt: [512 x 512] bf16, Bt[n*512+k]=W[k][n]
// one wave -> 32x64 output strip (2 row-tiles x 4 col-tiles), K-loop step 32
// per K-step: 4 A b128 loads + 8 B b128 loads -> 8 WMMAs (1.5 loads/wmma)
__global__ __launch_bounds__(128) void wmma_gemm_bf16_kernel(
        const __bf16* __restrict__ A, const __bf16* __restrict__ Bt,
        const float* __restrict__ bias, float* __restrict__ C, int M) {
    const int wave = (int)((blockIdx.x * blockDim.x + threadIdx.x) >> 5);
    const int lane = threadIdx.x & 31;
    const int tiles_n = HH / 64;                     // 8 strips of 64 cols
    const int tm = wave / tiles_n;                   // 32-row tile
    const int tn = wave % tiles_n;                   // 64-col strip
    if (tm * 32 >= M) return;

    const int l15   = lane & 15;
    const int khalf = (lane >> 4) * 8;               // ISA 16-bit A layout
    const int row0  = tm * 32 + l15;
    const int row1  = row0 + 16;

    v8f acc[2][4] = {{v8f{}, v8f{}, v8f{}, v8f{}},
                     {v8f{}, v8f{}, v8f{}, v8f{}}};

    for (int k0 = 0; k0 < HH; k0 += 32) {
        const __bf16* ap0 = A + (long)row0 * HH + k0 + khalf;
        const __bf16* ap1 = A + (long)row1 * HH + k0 + khalf;
        v8bf a0lo = *(const v8bf*)(ap0);
        v8bf a0hi = *(const v8bf*)(ap0 + 16);
        v8bf a1lo = *(const v8bf*)(ap1);
        v8bf a1hi = *(const v8bf*)(ap1 + 16);
        v16bf a0 = __builtin_shufflevector(a0lo, a0hi,
                     0,1,2,3,4,5,6,7,8,9,10,11,12,13,14,15);
        v16bf a1 = __builtin_shufflevector(a1lo, a1hi,
                     0,1,2,3,4,5,6,7,8,9,10,11,12,13,14,15);
        #pragma unroll
        for (int j = 0; j < 4; ++j) {
            const int col = tn * 64 + j * 16 + l15;
            const __bf16* bp = Bt + (long)col * HH + k0 + khalf;
            v8bf blo = *(const v8bf*)(bp);
            v8bf bhi = *(const v8bf*)(bp + 16);
            v16bf b = __builtin_shufflevector(blo, bhi,
                        0,1,2,3,4,5,6,7,8,9,10,11,12,13,14,15);
            acc[0][j] = __builtin_amdgcn_wmma_f32_16x16x32_bf16(
                            false, a0, false, b, (short)0, acc[0][j], false, false);
            acc[1][j] = __builtin_amdgcn_wmma_f32_16x16x32_bf16(
                            false, a1, false, b, (short)0, acc[1][j], false, false);
        }
    }

    // C/D layout: VGPR r, lanes 0-15 -> M=r, lanes 16-31 -> M=8+r; N = lane&15
    #pragma unroll
    for (int i = 0; i < 2; ++i) {
        const int mbase = tm * 32 + i * 16 + (lane >> 4) * 8;
        #pragma unroll
        for (int j = 0; j < 4; ++j) {
            const int n = tn * 64 + j * 16 + l15;
            const float bv = bias[n];
            #pragma unroll
            for (int r = 0; r < 8; ++r)
                C[(long)(mbase + r) * HH + n] = acc[i][j][r] + bv;
        }
    }
}

// ----------------------------------- per-feature mean / biased variance
__global__ __launch_bounds__(256) void colstats_kernel(
        const float* __restrict__ z, float* __restrict__ mu,
        float* __restrict__ var) {
    __shared__ float ssum[256], ssq[256];
    const int f = blockIdx.x;                        // feature column
    float s = 0.0f, q = 0.0f;
    for (int n = threadIdx.x; n < NN; n += 256) {
        float v = z[(long)n * HH + f];
        s += v; q += v * v;
    }
    ssum[threadIdx.x] = s; ssq[threadIdx.x] = q;
    __syncthreads();
    for (int off = 128; off > 0; off >>= 1) {
        if (threadIdx.x < off) {
            ssum[threadIdx.x] += ssum[threadIdx.x + off];
            ssq[threadIdx.x]  += ssq[threadIdx.x + off];
        }
        __syncthreads();
    }
    if (threadIdx.x == 0) {
        float m = ssum[0] / (float)NN;
        mu[f]  = m;
        var[f] = ssq[0] / (float)NN - m * m;
    }
}

// ------------------------------------------------- BN(train) + ReLU in place
__global__ void bnrelu_kernel(float* __restrict__ z,
                              const float* __restrict__ mu,
                              const float* __restrict__ var,
                              const float* __restrict__ gamma,
                              const float* __restrict__ beta) {
    long i = (long)blockIdx.x * blockDim.x + threadIdx.x;
    if (i >= (long)NN * HH) return;
    int f = (int)(i % HH);
    float v = gamma[f] * (z[i] - mu[f]) * rsqrtf(var[f] + BN_EPS) + beta[f];
    z[i] = fmaxf(v, 0.0f);
}

// --------------------------------------------- per-graph mean-nodes readout
__global__ void readout_accum_kernel(const float* __restrict__ h,
                                     const int* __restrict__ gids,
                                     float* __restrict__ femb, int off) {
    long i = (long)blockIdx.x * blockDim.x + threadIdx.x;
    if (i >= (long)NN * HH) return;
    int n = (int)(i / HH), f = (int)(i % HH);
    atomicAdd(&femb[(long)gids[n] * (2 * HH) + off + f], h[i]);
}
__global__ void readout_cnt_kernel(const int* __restrict__ gids,
                                   float* __restrict__ cnt) {
    int n = blockIdx.x * blockDim.x + threadIdx.x;
    if (n < NN) atomicAdd(&cnt[gids[n]], 1.0f);
}
__global__ void readout_div_kernel(float* __restrict__ femb,
                                   const float* __restrict__ cnt, int off) {
    int i = blockIdx.x * blockDim.x + threadIdx.x;
    if (i >= BB * HH) return;
    int b = i / HH, f = i % HH;
    femb[(long)b * (2 * HH) + off + f] /= fmaxf(cnt[b], 1.0f);
}

// --------------------------------------------------------------- MLP head
__global__ void fc1_kernel(const float* __restrict__ femb,   // [BB, 2H]
                           const float* __restrict__ Wfc,    // [2H, 256]
                           const float* __restrict__ bfc,
                           float* __restrict__ o1) {         // [BB, 256]
    int i = blockIdx.x * blockDim.x + threadIdx.x;
    if (i >= BB * 256) return;
    int b = i / 256, j = i % 256;
    const float* xr = femb + (long)b * (2 * HH);
    float s = bfc[j];
    for (int k = 0; k < 2 * HH; ++k) s += xr[k] * Wfc[(long)k * 256 + j];
    o1[i] = fmaxf(s, 0.0f);
}
__global__ void fc2_kernel(const float* __restrict__ o1,
                           const float* __restrict__ Wfc2,   // [256]
                           const float* __restrict__ bfc2,
                           float* __restrict__ out) {        // [BB]
    int b = blockIdx.x * blockDim.x + threadIdx.x;
    if (b >= BB) return;
    float s = bfc2[0];
    const float* xr = o1 + (long)b * 256;
    for (int j = 0; j < 256; ++j) s += xr[j] * Wfc2[j];
    out[b] = 1.0f / (1.0f + expf(-s));
}

// ================================================================= launch
static inline int blocks_for(long n, int bs) { return (int)((n + bs - 1) / bs); }

extern "C" void kernel_launch(void* const* d_in, const int* in_sizes, int n_in,
                              void* d_out, int out_size, void* d_ws, size_t ws_size,
                              hipStream_t stream) {
    const float* x1    = (const float*)d_in[0];
    const float* x2    = (const float*)d_in[1];
    const int*   src1  = (const int*)d_in[2];
    const int*   dst1  = (const int*)d_in[3];
    const int*   src2  = (const int*)d_in[4];
    const int*   dst2  = (const int*)d_in[5];
    const int*   gids1 = (const int*)d_in[6];
    const int*   gids2 = (const int*)d_in[7];
    const float* Winit = (const float*)d_in[8];
    const float* binit = (const float*)d_in[9];
    const float* W1    = (const float*)d_in[10];
    const float* b1    = (const float*)d_in[11];
    const float* W2    = (const float*)d_in[12];
    const float* b2    = (const float*)d_in[13];
    const float* g1    = (const float*)d_in[14];
    const float* be1   = (const float*)d_in[15];
    const float* g2    = (const float*)d_in[16];
    const float* be2   = (const float*)d_in[17];
    const float* Wfc   = (const float*)d_in[18];
    const float* bfc   = (const float*)d_in[19];
    const float* Wfc2  = (const float*)d_in[20];
    const float* bfc2  = (const float*)d_in[21];
    float* out = (float*)d_out;

    // ---- workspace carve-up
    char* ws = (char*)d_ws;
    const long NH = (long)NN * HH;
    float*  h    = (float*)ws;                 ws += NH * 4;            // 64MB
    float*  agg  = (float*)ws;                 ws += NH * 4;            // 64MB
    __bf16* ubf  = (__bf16*)ws;                ws += NH * 2;            // 32MB
    __bf16* W1t  = (__bf16*)ws;                ws += (long)HH * HH * 2; // .5MB
    __bf16* W2t  = (__bf16*)ws;                ws += (long)HH * HH * 2; // .5MB
    float*  deg  = (float*)ws;                 ws += (long)NN * 4;
    float*  mu   = (float*)ws;                 ws += HH * 4;
    float*  var  = (float*)ws;                 ws += HH * 4;
    float*  femb = (float*)ws;                 ws += (long)BB * 2 * HH * 4;
    float*  cnt  = (float*)ws;                 ws += BB * 4;
    float*  o1   = (float*)ws;                 ws += (long)BB * 256 * 4;

    const int T = 256;
    const long HW = (long)HH * HH;

    // ---- one-time weight transpose+convert (bf16)
    transpose_w_bf16_kernel<<<blocks_for(HW, T), T, 0, stream>>>(W1, W1t);
    transpose_w_bf16_kernel<<<blocks_for(HW, T), T, 0, stream>>>(W2, W2t);
    fill_zero_kernel<<<blocks_for((long)BB * 2 * HH, T), T, 0, stream>>>(femb, (long)BB * 2 * HH);

    const int gemm_waves  = (NN / 32) * (HH / 64);          // 8192 waves
    const int gemm_blocks = (gemm_waves * 32) / 128;        // 2048 blocks

    for (int g = 0; g < 2; ++g) {
        const float* x    = g ? x2 : x1;
        const int*   src  = g ? src2 : src1;
        const int*   dst  = g ? dst2 : dst1;
        const int*   gids = g ? gids2 : gids1;
        const int    off  = g * HH;

        // h0 = x @ Winit + binit
        init_linear_kernel<<<blocks_for(NH, T), T, 0, stream>>>(x, Winit, binit, h);

        // degree (per-graph, reused by both layers)
        fill_zero_kernel<<<blocks_for(NN, T), T, 0, stream>>>(deg, NN);
        deg_kernel<<<blocks_for(EE, T), T, 0, stream>>>(dst, deg);

        const float* Wb[2]    = {b1, b2};
        const __bf16* Wt[2]   = {W1t, W2t};
        const float* Gam[2]   = {g1, g2};
        const float* Bet[2]   = {be1, be2};

        for (int layer = 0; layer < 2; ++layer) {
            // mean aggregation: agg = segment_sum(h[src], dst) / max(deg,1)
            fill_zero_kernel<<<blocks_for(NH, T), T, 0, stream>>>(agg, NH);
            scatter_kernel<<<blocks_for((long)EE * (HH / 4), T), T, 0, stream>>>(h, src, dst, agg);
            combine_bf16_kernel<<<blocks_for(NH, T), T, 0, stream>>>(h, agg, deg, ubf);

            // z = u @ W + b  (bf16 WMMA, f32 accumulate) -> overwrites h
            wmma_gemm_bf16_kernel<<<gemm_blocks, 128, 0, stream>>>(ubf, Wt[layer], Wb[layer], h, NN);

            // BN(train) + ReLU
            colstats_kernel<<<HH, 256, 0, stream>>>(h, mu, var);
            bnrelu_kernel<<<blocks_for(NH, T), T, 0, stream>>>(h, mu, var, Gam[layer], Bet[layer]);
        }

        // mean_nodes readout into concatenated [BB, 2H]
        fill_zero_kernel<<<blocks_for(BB, T), T, 0, stream>>>(cnt, BB);
        readout_accum_kernel<<<blocks_for(NH, T), T, 0, stream>>>(h, gids, femb, off);
        readout_cnt_kernel<<<blocks_for(NN, T), T, 0, stream>>>(gids, cnt);
        readout_div_kernel<<<blocks_for((long)BB * HH, T), T, 0, stream>>>(femb, cnt, off);
    }

    // MLP head
    fc1_kernel<<<blocks_for((long)BB * 256, T), T, 0, stream>>>(femb, Wfc, bfc, o1);
    fc2_kernel<<<1, BB, 0, stream>>>(o1, Wfc2, bfc2, out);
}
